// DAM_79774722556285
// MI455X (gfx1250) — matrix operand order repaired and checked by reading, not compiled, not run on hardware
//
#include <hip/hip_runtime.h>

// ---------------------------------------------------------------------------
// MI455X (gfx1250) dual-attention module:
//   F = Z^T W^T + b ; S = softmax(F F^T) ; R = softmax(F^T F)
//   S2 = F S ; Out = S2 + R S2      (per batch, all matrices 1024x1024)
// 172 GFLOP of GEMM vs ~250MB traffic -> compute bound -> bf16 WMMA
// (v_wmma_f32_16x16x32_bf16), fp32 accumulate, fp32 softmax logits.
// v2: 64x64 wave tiles (16 wmma / K-step), double-buffered LDS fed by
//     GLOBAL_LOAD_ASYNC_TO_LDS_B128 (ASYNCcnt) when the builtin exists.
// ---------------------------------------------------------------------------

typedef __bf16 bf16;
typedef __attribute__((ext_vector_type(16))) __bf16 v16bf;
typedef __attribute__((ext_vector_type(8)))  __bf16 bf16x8;
typedef __attribute__((ext_vector_type(8)))  float  v8f;

#define DIM 1024
#define BK  32
#define BM  128
#define BN  256
#define LDT 40   // LDS row stride (elems): keeps 16B alignment, rotates banks

#if defined(__HIP_DEVICE_COMPILE__) && defined(__has_builtin)
#if __has_builtin(__builtin_amdgcn_global_load_async_to_lds_b128) && \
    __has_builtin(__builtin_amdgcn_s_wait_asynccnt)
#define USE_ASYNC_LDS 1
#endif
#endif
#ifndef USE_ASYNC_LDS
#define USE_ASYNC_LDS 0
#endif

#if USE_ASYNC_LDS
// 16B global -> LDS DMA (no VGPR round-trip).
// Builtin prototype (from hipcc diagnostic):
//   void (...)(int __vector(4) addrspace(1)*, int __vector(4) addrspace(3)*,
//              imm int offset, imm int cpol)
// Address-space pointers are built via integer casts (generic LDS address's
// low 32 bits == LDS byte offset per the CDNA5 aperture rules).
typedef int v4i_ __attribute__((vector_size(16)));
typedef __attribute__((address_space(1))) v4i_ g_v4i;
typedef __attribute__((address_space(3))) v4i_ l_v4i;

__device__ __forceinline__ void async_cp16(const bf16* g, bf16* l)
{
    g_v4i* gp = (g_v4i*)(unsigned long long)(uintptr_t)g;
    l_v4i* lp = (l_v4i*)(unsigned int)(uintptr_t)l;
    __builtin_amdgcn_global_load_async_to_lds_b128(gp, lp, 0, 0);
}
#endif

// ---------------------------------------------------------------------------
// Generic 1024x1024x1024 GEMM:  C[m][n] = sum_k A[m][k] * Bt[n][k]
// mode 0: +bias[n], store bf16 C and bf16 C^T        (produces F, F^T)
// mode 1: store fp32 C                               (produces Gs / Gr)
// mode 2: store fp32 C and bf16 C^T                  (produces S2, S2^T)
// mode 3: store fp32 (C + addend)                    (final output)
// Block tile 128(M) x 256(N), 256 threads = 8 waves (2x4), wave tile 64x64.
// ---------------------------------------------------------------------------
__global__ __launch_bounds__(256) void dam_wmma_gemm(
    const bf16* __restrict__ A, const bf16* __restrict__ Bt,
    const float* __restrict__ bias, const float* __restrict__ addend,
    float* __restrict__ Cf, bf16* __restrict__ Cbf, bf16* __restrict__ Cbft,
    int mode)
{
    __shared__ __attribute__((aligned(16))) bf16 sA[2][BM * LDT];
    __shared__ __attribute__((aligned(16))) bf16 sB[2][BN * LDT];

    const int t    = threadIdx.x;
    const int bm   = blockIdx.y * BM;
    const int bn   = blockIdx.x * BN;
    const int wid  = t >> 5;
    const int lane = t & 31;
    const int wm   = (wid >> 2) * 64;   // 2 wave-rows of 64
    const int wn   = (wid & 3) * 64;    // 4 wave-cols of 64
    const int lr   = lane & 15;
    const int hi   = (lane >> 4);
    const int kbA  = hi * 8;            // A frag: K kb..kb+7 and kb+16..kb+23
    const int kbB  = hi * 8;            // B frag: K kb..kb+7 and kb+16..kb+23

    const v8f vzero = {0.f, 0.f, 0.f, 0.f, 0.f, 0.f, 0.f, 0.f};
    v8f acc[4][4];
#pragma unroll
    for (int mi = 0; mi < 4; mi++)
#pragma unroll
        for (int ni = 0; ni < 4; ni++) acc[mi][ni] = vzero;

#if USE_ASYNC_LDS
    auto load_tile_async = [&](int k0, int buf) {
#pragma unroll
        for (int i = 0; i < 2; i++) {          // A: 128x32 = 512 x 16B chunks
            int c = t + 256 * i, row = c >> 2, kc = (c & 3) * 8;
            async_cp16(&A[(size_t)(bm + row) * DIM + k0 + kc],
                       &sA[buf][row * LDT + kc]);
        }
#pragma unroll
        for (int i = 0; i < 4; i++) {          // B: 256x32 = 1024 x 16B chunks
            int c = t + 256 * i, row = c >> 2, kc = (c & 3) * 8;
            async_cp16(&Bt[(size_t)(bn + row) * DIM + k0 + kc],
                       &sB[buf][row * LDT + kc]);
        }
    };
#endif

    auto compute_step = [&](int buf) {
        v16bf afrag[4], bfrag[4];
#pragma unroll
        for (int mi = 0; mi < 4; mi++) {
            const bf16* p = &sA[buf][(wm + mi * 16 + lr) * LDT + kbA];
            bf16x8 lo = *reinterpret_cast<const bf16x8*>(p);
            bf16x8 ho = *reinterpret_cast<const bf16x8*>(p + 16);
#pragma unroll
            for (int e = 0; e < 8; e++) { afrag[mi][e] = lo[e]; afrag[mi][8 + e] = ho[e]; }
        }
#pragma unroll
        for (int ni = 0; ni < 4; ni++) {
            const bf16* p = &sB[buf][(wn + ni * 16 + lr) * LDT + kbB];
            bf16x8 lo = *reinterpret_cast<const bf16x8*>(p);
            bf16x8 ho = *reinterpret_cast<const bf16x8*>(p + 16);
#pragma unroll
            for (int e = 0; e < 8; e++) { bfrag[ni][e] = lo[e]; bfrag[ni][8 + e] = ho[e]; }
        }
#pragma unroll
        for (int mi = 0; mi < 4; mi++)
#pragma unroll
            for (int ni = 0; ni < 4; ni++)
                acc[mi][ni] = __builtin_amdgcn_wmma_f32_16x16x32_bf16(
                    false, afrag[mi], false, bfrag[ni],
                    (short)0, acc[mi][ni], false, false);
    };

#if USE_ASYNC_LDS
    // software pipeline: DMA tile k+1 into the other buffer while WMMAs
    // chew on tile k; one barrier + one s_wait_asynccnt per K-step.
    load_tile_async(0, 0);
    for (int k0 = 0; k0 < DIM; k0 += BK) {
        const int cur = (k0 / BK) & 1;
        __builtin_amdgcn_s_wait_asynccnt(0);   // own DMAs for 'cur' done
        __syncthreads();                        // everyone's done + WAR on nxt
        if (k0 + BK < DIM) load_tile_async(k0 + BK, cur ^ 1);
        compute_step(cur);
    }
    __syncthreads();
#else
    // fallback: synchronous global->VGPR->LDS staging, single buffer
    for (int k0 = 0; k0 < DIM; k0 += BK) {
        bf16x8 ra[2], rb[4];
#pragma unroll
        for (int i = 0; i < 2; i++) {
            int c = t + 256 * i, row = c >> 2, kc = (c & 3) * 8;
            ra[i] = *reinterpret_cast<const bf16x8*>(&A[(size_t)(bm + row) * DIM + k0 + kc]);
        }
#pragma unroll
        for (int i = 0; i < 4; i++) {
            int c = t + 256 * i, row = c >> 2, kc = (c & 3) * 8;
            rb[i] = *reinterpret_cast<const bf16x8*>(&Bt[(size_t)(bn + row) * DIM + k0 + kc]);
        }
        __syncthreads();                        // WAR: prev compute finished
#pragma unroll
        for (int i = 0; i < 2; i++) {
            int c = t + 256 * i, row = c >> 2, kc = (c & 3) * 8;
            *reinterpret_cast<bf16x8*>(&sA[0][row * LDT + kc]) = ra[i];
        }
#pragma unroll
        for (int i = 0; i < 4; i++) {
            int c = t + 256 * i, row = c >> 2, kc = (c & 3) * 8;
            *reinterpret_cast<bf16x8*>(&sB[0][row * LDT + kc]) = rb[i];
        }
        __syncthreads();
        compute_step(0);
    }
#endif

    // ---- epilogue ----
    const int mbase = bm + wm, nbase = bn + wn;
#pragma unroll
    for (int mi = 0; mi < 4; mi++) {
#pragma unroll
        for (int ni = 0; ni < 4; ni++) {
            int col = nbase + ni * 16 + lr;
            float bv = (mode == 0) ? bias[col] : 0.f;
#pragma unroll
            for (int e = 0; e < 8; e++) {
                int row = mbase + mi * 16 + e + hi * 8;
                float v = acc[mi][ni][e];
                size_t idx  = (size_t)row * DIM + col;
                size_t idxt = (size_t)col * DIM + row;
                if (mode == 0) {
                    v += bv;
                    Cbf[idx]   = (bf16)v;
                    Cbft[idxt] = (bf16)v;
                } else if (mode == 1) {
                    Cf[idx] = v;
                } else if (mode == 2) {
                    Cf[idx] = v;
                    Cbft[idxt] = (bf16)v;
                } else {
                    Cf[idx] = v + addend[idx];
                }
            }
        }
    }
}

// ---------------------------------------------------------------------------
// Row softmax of a 1024x1024 fp32 matrix -> bf16 (optionally transposed write)
// ---------------------------------------------------------------------------
__global__ __launch_bounds__(256) void dam_softmax_row(
    const float* __restrict__ G, bf16* __restrict__ out, int transposed)
{
    __shared__ float red[256];
    const int r = blockIdx.x, t = threadIdx.x;
    const float* row = G + (size_t)r * DIM;

    float x[4];
    float mx = -3.4e38f;
#pragma unroll
    for (int i = 0; i < 4; i++) { x[i] = row[t + 256 * i]; mx = fmaxf(mx, x[i]); }
    red[t] = mx; __syncthreads();
    for (int s = 128; s > 0; s >>= 1) {
        if (t < s) red[t] = fmaxf(red[t], red[t + s]);
        __syncthreads();
    }
    mx = red[0]; __syncthreads();

    float sum = 0.f;
#pragma unroll
    for (int i = 0; i < 4; i++) { x[i] = __expf(x[i] - mx); sum += x[i]; }
    red[t] = sum; __syncthreads();
    for (int s = 128; s > 0; s >>= 1) {
        if (t < s) red[t] += red[t + s];
        __syncthreads();
    }
    float inv = 1.0f / red[0];

#pragma unroll
    for (int i = 0; i < 4; i++) {
        int c = t + 256 * i;
        bf16 v = (bf16)(x[i] * inv);
        if (transposed) out[(size_t)c * DIM + r] = v;
        else            out[(size_t)r * DIM + c] = v;
    }
}

// ---------------------------------------------------------------------------
// 1024x1024 fp32 -> bf16 transpose (LDS tiled):  out[p][k] = in[k][p]
// ---------------------------------------------------------------------------
__global__ __launch_bounds__(256) void dam_transpose_cvt(
    const float* __restrict__ in, bf16* __restrict__ out)
{
    __shared__ float tile[32][33];
    const int tx = threadIdx.x, ty = threadIdx.y;  // block (32,8)
    int x = blockIdx.x * 32 + tx;
#pragma unroll
    for (int j = 0; j < 4; j++) {
        int y = blockIdx.y * 32 + ty + j * 8;
        tile[ty + j * 8][tx] = in[(size_t)y * DIM + x];
    }
    __syncthreads();
    int x2 = blockIdx.y * 32 + tx;
#pragma unroll
    for (int j = 0; j < 4; j++) {
        int y2 = blockIdx.x * 32 + ty + j * 8;
        out[(size_t)y2 * DIM + x2] = (bf16)tile[tx][ty + j * 8];
    }
}

// fp32 -> bf16 elementwise (for W, shared across batches)
__global__ __launch_bounds__(256) void dam_cvt_bf16(
    const float* __restrict__ in, bf16* __restrict__ out, int n)
{
    int i = blockIdx.x * 256 + threadIdx.x;
    if (i < n) out[i] = (bf16)in[i];
}

// ---------------------------------------------------------------------------
extern "C" void kernel_launch(void* const* d_in, const int* in_sizes, int n_in,
                              void* d_out, int out_size, void* d_ws, size_t ws_size,
                              hipStream_t stream)
{
    const float* z = (const float*)d_in[0];   // (16, 1024, 32, 32)
    const float* W = (const float*)d_in[1];   // (1024, 1024)
    const float* b = (const float*)d_in[2];   // (1024,)
    float* out = (float*)d_out;               // (16, 1024, 1024) fp32

    char* ws = (char*)d_ws;
    const size_t MB = 1u << 20;
    bf16*  W_bf  = (bf16*)(ws +  0 * MB);   // 2MB  W (row-major = Bt for GEMM1)
    bf16*  Zt    = (bf16*)(ws +  2 * MB);   // 2MB  Z^T bf16
    bf16*  F_bf  = (bf16*)(ws +  4 * MB);   // 2MB  F
    bf16*  Ft_bf = (bf16*)(ws +  6 * MB);   // 2MB  F^T
    float* G     = (float*)(ws +  8 * MB);  // 4MB  softmax logits (reused)
    bf16*  St    = (bf16*)(ws + 12 * MB);   // 2MB  S^T
    bf16*  R_bf  = (bf16*)(ws + 14 * MB);   // 2MB  R
    float* S2    = (float*)(ws + 16 * MB);  // 4MB  S2 fp32
    bf16*  S2t   = (bf16*)(ws + 20 * MB);   // 2MB  S2^T
    (void)in_sizes; (void)n_in; (void)out_size; (void)ws_size;

    const dim3 gG(DIM / BN, DIM / BM), bG(256);   // (4, 8) x 256
    const dim3 gT(32, 32), bT(32, 8);

    // W -> bf16 once (B-operand of GEMM1 is W^T, supplied N-major == W rows)
    dam_cvt_bf16<<<4096, 256, 0, stream>>>(W, W_bf, DIM * DIM);

    for (int n = 0; n < 16; n++) {
        const float* zb = z + (size_t)n * DIM * DIM;
        // Z^T in bf16
        dam_transpose_cvt<<<gT, bT, 0, stream>>>(zb, Zt);
        // F = Z^T W^T + b   -> F_bf, Ft_bf
        dam_wmma_gemm<<<gG, bG, 0, stream>>>(Zt, W_bf, b, nullptr,
                                             nullptr, F_bf, Ft_bf, 0);
        // Gs = F F^T -> softmax rows -> S^T (bf16)
        dam_wmma_gemm<<<gG, bG, 0, stream>>>(F_bf, F_bf, nullptr, nullptr,
                                             G, nullptr, nullptr, 1);
        dam_softmax_row<<<DIM, 256, 0, stream>>>(G, St, 1);
        // Gr = F^T F -> softmax rows -> R (bf16)
        dam_wmma_gemm<<<gG, bG, 0, stream>>>(Ft_bf, Ft_bf, nullptr, nullptr,
                                             G, nullptr, nullptr, 1);
        dam_softmax_row<<<DIM, 256, 0, stream>>>(G, R_bf, 0);
        // S2 = F S  (Bt = S^T) -> S2 fp32 + S2^T bf16
        dam_wmma_gemm<<<gG, bG, 0, stream>>>(F_bf, St, nullptr, nullptr,
                                             S2, nullptr, S2t, 2);
        // Out = S2 + R S2  (Bt = S2^T, addend = S2)
        dam_wmma_gemm<<<gG, bG, 0, stream>>>(R_bf, S2t, nullptr, S2,
                                             out + (size_t)n * DIM * DIM,
                                             nullptr, nullptr, 3);
    }
}
